// EdgeGAT_2937757630806
// MI455X (gfx1250) — matrix operand order repaired and checked by reading, not compiled, not run on hardware
//
#include <hip/hip_runtime.h>
#include <hip/hip_bf16.h>
#include <math.h>

#define SLOPE 0.2f

typedef __attribute__((ext_vector_type(16))) __bf16 v16bf;
typedef __attribute__((ext_vector_type(8)))  float  v8f;

union ABf { v16bf v; unsigned int u[8]; };

__device__ __forceinline__ unsigned short f2bf(float f) {
  union { float f; unsigned int u; } c; c.f = f;
  unsigned int r = c.u + 0x7FFFu + ((c.u >> 16) & 1u);
  return (unsigned short)(r >> 16);
}

__device__ __forceinline__ void atomicMaxF(float* addr, float v) {
  // int/uint punning trick: correct for mixed-sign f32 max, native atomics
  if (v >= 0.0f) atomicMax((int*)addr, __float_as_int(v));
  else           atomicMin((unsigned int*)addr, __float_as_uint(v));
}

// ---------------- utility kernels ----------------

__global__ void fill_k(float* p, float v, size_t n) {
  size_t i = (size_t)blockIdx.x * blockDim.x + threadIdx.x;
  if (i < n) p[i] = v;
}

__global__ void f2bf_k(const float* in, unsigned short* out, size_t n) {
  size_t i = (size_t)blockIdx.x * blockDim.x + threadIdx.x;
  if (i < n) out[i] = f2bf(in[i]);
}

// out[n*Kpad + k] = bf16(W[k*Nc + n]), zero-padded for k >= K
__global__ void pack_wt_k(const float* W, unsigned short* out, int K, int Nc, int Kpad) {
  int i = blockIdx.x * blockDim.x + threadIdx.x;
  if (i >= Nc * Kpad) return;
  int n = i / Kpad, k = i % Kpad;
  float v = (k < K) ? W[(size_t)k * Nc + n] : 0.0f;
  out[i] = f2bf(v);
}

// ---------------- WMMA GEMM: C[M,Nc] = A[M,K] @ Bt[Nc,K]^T ----------------
// A bf16 row-major [M,K]; Bt bf16 row-major [Nc,K] (W pre-transposed).
__global__ void wmma_gemm_k(const unsigned short* __restrict__ A,
                            const unsigned short* __restrict__ Bt,
                            float* __restrict__ C, int M, int K, int Nc) {
  int wave  = blockIdx.x * 8 + (threadIdx.x >> 5);
  int tilesM = (M + 15) >> 4;
  int tilesN = Nc >> 4;
  if (wave >= tilesM * tilesN) return;          // wave-uniform guard
  int tm = wave / tilesN, tn = wave % tilesN;
  int lane = threadIdx.x & 31;
  int hx = lane >> 4;                            // half-wave 0/1
  int lm = lane & 15;
  int mrow = tm * 16 + lm; if (mrow >= M) mrow = M - 1;
  const unsigned int* A32 = (const unsigned int*)(A + (size_t)mrow * K);
  const unsigned int* B32 = (const unsigned int*)(Bt + (size_t)(tn * 16 + lm) * K);
  v8f c = {};
  for (int k0 = 0; k0 < K; k0 += 32) {
    ABf a, b;
#pragma unroll
    for (int r = 0; r < 8; ++r) {
      int ka = k0 + ((r < 4) ? 0 : 16) + hx * 8 + (r & 3) * 2;   // A frag layout
      a.u[r] = A32[ka >> 1];
      int kb = k0 + ((r < 4) ? 0 : 16) + hx * 8 + (r & 3) * 2;   // B frag layout
      b.u[r] = B32[kb >> 1];
    }
    c = __builtin_amdgcn_wmma_f32_16x16x32_bf16(false, a.v, false, b.v,
                                                (short)0, c, false, false);
  }
  // C layout: n = lane&15, m = r + 8*half
#pragma unroll
  for (int r = 0; r < 8; ++r) {
    int m = tm * 16 + r + hx * 8;
    if (m < M) C[(size_t)m * Nc + tn * 16 + lm] = c[r];
  }
}

// ---------------- attention per-node dots: ls/ld [N,H] ----------------
__global__ void attn_k(const float* __restrict__ hfeat, const float* __restrict__ as_,
                       const float* __restrict__ ad_, float* __restrict__ ls,
                       float* __restrict__ ld, int N, int H) {
  int wid = blockIdx.x * 8 + (threadIdx.x >> 5);
  if (wid >= N * H) return;
  int lane = threadIdx.x & 31;
  int n = wid / H, h = wid % H;
  const float* row = hfeat + (size_t)n * H * 64 + h * 64;
  float v0 = row[lane], v1 = row[lane + 32];
  float ps = v0 * as_[h * 64 + lane] + v1 * as_[h * 64 + lane + 32];
  float pd = v0 * ad_[h * 64 + lane] + v1 * ad_[h * 64 + lane + 32];
#pragma unroll
  for (int o = 16; o > 0; o >>= 1) {
    ps += __shfl_xor(ps, o, 32);
    pd += __shfl_xor(pd, o, 32);
  }
  if (lane == 0) { ls[wid] = ps; ld[wid] = pd; }
}

// ---------------- edge passes (E' = E + N self-loops) ----------------
__global__ void edge_max_k(const long long* __restrict__ ei, int E, int N,
                           const float* __restrict__ ls, const float* __restrict__ ld,
                           float* __restrict__ mseg, int H) {
  int e = blockIdx.x * blockDim.x + threadIdx.x;
  if (e >= E + N) return;
  int s = (e < E) ? (int)ei[e] : (e - E);
  int d = (e < E) ? (int)ei[E + e] : (e - E);
  for (int h = 0; h < H; ++h) {
    float x = ls[s * H + h] + ld[d * H + h];
    x = (x > 0.0f) ? x : SLOPE * x;
    atomicMaxF(&mseg[d * H + h], x);
  }
}

__global__ void edge_expsum_k(const long long* __restrict__ ei, int E, int N,
                              const float* __restrict__ ls, const float* __restrict__ ld,
                              const float* __restrict__ mseg, float* __restrict__ sseg, int H) {
  int e = blockIdx.x * blockDim.x + threadIdx.x;
  if (e >= E + N) return;
  int s = (e < E) ? (int)ei[e] : (e - E);
  int d = (e < E) ? (int)ei[E + e] : (e - E);
  for (int h = 0; h < H; ++h) {
    float x = ls[s * H + h] + ld[d * H + h];
    x = (x > 0.0f) ? x : SLOPE * x;
    float mm = mseg[d * H + h];
    if (mm < -1e30f) mm = 0.0f;
    atomicAdd(&sseg[d * H + h], __expf(x - mm));
  }
}

// 64 threads per (edge, head): coalesced gather + f32 atomic scatter-add
__global__ void edge_scatter_k(const long long* __restrict__ ei, int E, int N,
                               const float* __restrict__ ls, const float* __restrict__ ld,
                               const float* __restrict__ mseg, const float* __restrict__ sseg,
                               const float* __restrict__ hfeat, float* __restrict__ agg, int H) {
  long long g = (long long)blockIdx.x * blockDim.x + threadIdx.x;
  long long grp = g >> 6;
  int dch = (int)(g & 63);
  long long tot = (long long)(E + N) * H;
  if (grp >= tot) return;
  int h = (int)(grp % H);
  long long e = grp / H;
  int s = (e < E) ? (int)ei[e] : (int)(e - E);
  int d = (e < E) ? (int)ei[E + e] : (int)(e - E);
  float x = ls[s * H + h] + ld[d * H + h];
  x = (x > 0.0f) ? x : SLOPE * x;
  float mm = mseg[d * H + h];
  if (mm < -1e30f) mm = 0.0f;
  float alpha = __expf(x - mm) / (sseg[d * H + h] + 1e-16f);
  int HD = H * 64;
  atomicAdd(&agg[(size_t)d * HD + h * 64 + dch],
            hfeat[(size_t)s * HD + h * 64 + dch] * alpha);
}

// ---------------- bias (+optional ELU) ----------------
__global__ void bias_act_k(const float* __restrict__ in, const float* __restrict__ b,
                           float* __restrict__ out, size_t total, int C, int doElu) {
  size_t i = (size_t)blockIdx.x * blockDim.x + threadIdx.x;
  if (i >= total) return;
  float v = in[i] + b[i % C];
  if (doElu) v = (v > 0.0f) ? v : (__expf(v) - 1.0f);
  out[i] = v;
}

// ---------------- batch-norm ----------------
__global__ void bn_stats_k(const float* __restrict__ y, float* __restrict__ mu,
                           float* __restrict__ var, int N, int C) {
  __shared__ float sh1[256], sh2[256];
  int c = blockIdx.x, t = threadIdx.x;
  float s = 0.0f, s2 = 0.0f;
  for (int n = t; n < N; n += blockDim.x) {
    float v = y[(size_t)n * C + c];
    s += v; s2 += v * v;
  }
  sh1[t] = s; sh2[t] = s2; __syncthreads();
  for (int o = 128; o > 0; o >>= 1) {
    if (t < o) { sh1[t] += sh1[t + o]; sh2[t] += sh2[t + o]; }
    __syncthreads();
  }
  if (t == 0) {
    float m_ = sh1[0] / N;
    mu[c] = m_; var[c] = sh2[0] / N - m_ * m_;
  }
}

// applies BN and emits bf16 copy (next layer's GEMM A operand)
__global__ void bn_apply_k(const float* __restrict__ y, const float* __restrict__ mu,
                           const float* __restrict__ var, const float* __restrict__ g,
                           const float* __restrict__ b, unsigned short* __restrict__ outbf,
                           size_t total, int C) {
  size_t i = (size_t)blockIdx.x * blockDim.x + threadIdx.x;
  if (i >= total) return;
  int c = (int)(i % C);
  float v = (y[i] - mu[c]) * rsqrtf(var[c] + 1e-5f) * g[c] + b[c];
  outbf[i] = f2bf(v);
}

// ---------------- fused final kernel ----------------
// Per wave: 16 edges. edge MLP (2 WMMA GEMMs, K padded 16->32) + fc dot.
#define FE_WAVES 8
__global__ void final_edge_k(const long long* __restrict__ ei, int E,
                             const float* __restrict__ ea,
                             const unsigned short* __restrict__ mw1t,   // [64][32] bf16
                             const float* __restrict__ mb1,
                             const unsigned short* __restrict__ mw2t,   // [64][64] bf16
                             const float* __restrict__ mb2,
                             const float* __restrict__ h3,              // [N,64] f32
                             const float* __restrict__ fcW,             // [192]
                             const float* __restrict__ fcb,
                             float* __restrict__ out) {
  __shared__ unsigned short hidLds[FE_WAVES][16][72];  // bf16 hidden tile per wave
  __shared__ float edot[FE_WAVES][16];
  int w = threadIdx.x >> 5;
  int lane = threadIdx.x & 31;
  int hx = lane >> 4, lm = lane & 15;
  int eb = (blockIdx.x * FE_WAVES + w) * 16;
  if (eb >= E) return;                                  // wave-uniform

  // A fragment from edge_attr row (K: 16 real + 16 zero pad)
  ABf a;
  {
    const float* row = ea + (size_t)(eb + lm) * 16;
#pragma unroll
    for (int r = 0; r < 8; ++r) {
      int k = ((r < 4) ? 0 : 16) + hx * 8 + (r & 3) * 2;
      unsigned int lo = 0, hi = 0;
      if (k < 16) { lo = f2bf(row[k]); hi = f2bf(row[k + 1]); }
      a.u[r] = lo | (hi << 16);
    }
  }
  // GEMM1: hidden = relu(ea @ mlpW1 + b1) -> LDS (bf16)
#pragma unroll
  for (int nt = 0; nt < 4; ++nt) {
    ABf b;
    const unsigned int* B32 = (const unsigned int*)(mw1t + (size_t)(nt * 16 + lm) * 32);
#pragma unroll
    for (int r = 0; r < 8; ++r) b.u[r] = B32[(((r < 4) ? 0 : 16) + hx * 8 + (r & 3) * 2) >> 1];
    v8f c = {};
    c = __builtin_amdgcn_wmma_f32_16x16x32_bf16(false, a.v, false, b.v,
                                                (short)0, c, false, false);
    float bias = mb1[nt * 16 + lm];
#pragma unroll
    for (int r = 0; r < 8; ++r) {
      float v = c[r] + bias;
      v = (v > 0.0f) ? v : 0.0f;
      hidLds[w][r + hx * 8][nt * 16 + lm] = f2bf(v);
    }
  }
  __syncthreads();
  // GEMM2: ef = hidden @ mlpW2 + b2; fold fc dot into C fragments
  float part[8];
#pragma unroll
  for (int r = 0; r < 8; ++r) part[r] = 0.0f;
#pragma unroll
  for (int nt = 0; nt < 4; ++nt) {
    v8f c = {};
    const unsigned int* hrow = (const unsigned int*)&hidLds[w][lm][0];
#pragma unroll
    for (int kt = 0; kt < 2; ++kt) {
      ABf a2, b2;
#pragma unroll
      for (int r = 0; r < 8; ++r) {
        int k = kt * 32 + ((r < 4) ? 0 : 16) + hx * 8 + (r & 3) * 2;
        a2.u[r] = hrow[k >> 1];
      }
      const unsigned int* B32 = (const unsigned int*)(mw2t + (size_t)(nt * 16 + lm) * 64);
#pragma unroll
      for (int r = 0; r < 8; ++r) b2.u[r] = B32[(kt * 32 + ((r < 4) ? 0 : 16) + hx * 8 + (r & 3) * 2) >> 1];
      c = __builtin_amdgcn_wmma_f32_16x16x32_bf16(false, a2.v, false, b2.v,
                                                  (short)0, c, false, false);
    }
    float fw = fcW[128 + nt * 16 + lm];
    float bias = mb2[nt * 16 + lm];
#pragma unroll
    for (int r = 0; r < 8; ++r) part[r] += (c[r] + bias) * fw;
  }
  // butterfly reduce within each 16-lane half (edge m = r + 8*half)
#pragma unroll
  for (int r = 0; r < 8; ++r) {
    float p = part[r];
    p += __shfl_xor(p, 1, 32);
    p += __shfl_xor(p, 2, 32);
    p += __shfl_xor(p, 4, 32);
    p += __shfl_xor(p, 8, 32);
    part[r] = p;
  }
  if (lm == 0) {
#pragma unroll
    for (int r = 0; r < 8; ++r) edot[w][hx * 8 + r] = part[r];
  }
  __syncthreads();
  // add h[src]/h[dst] fc dots
  if (lane < 16) {
    int e = eb + lane;
    int s = (int)ei[e], d = (int)ei[E + e];
    const float* hs = h3 + (size_t)s * 64;
    const float* hd = h3 + (size_t)d * 64;
    float acc = edot[w][lane] + fcb[0];
#pragma unroll 8
    for (int i = 0; i < 64; ++i) acc += hs[i] * fcW[i] + hd[i] * fcW[64 + i];
    out[e] = acc;
  }
}

// ---------------- host ----------------
extern "C" void kernel_launch(void* const* d_in, const int* in_sizes, int n_in,
                              void* d_out, int out_size, void* d_ws, size_t ws_size,
                              hipStream_t stream) {
  const float*     x   = (const float*)d_in[0];
  const long long* ei  = (const long long*)d_in[1];
  const float*     ea  = (const float*)d_in[2];
  const float*     W1  = (const float*)d_in[3];
  const float*     a1s = (const float*)d_in[4];
  const float*     a1d = (const float*)d_in[5];
  const float*     b1  = (const float*)d_in[6];
  const float*     W2  = (const float*)d_in[7];
  const float*     a2s = (const float*)d_in[8];
  const float*     a2d = (const float*)d_in[9];
  const float*     b2  = (const float*)d_in[10];
  const float*     W3  = (const float*)d_in[11];
  const float*     a3s = (const float*)d_in[12];
  const float*     a3d = (const float*)d_in[13];
  const float*     b3  = (const float*)d_in[14];
  const float*     bn1g = (const float*)d_in[15];
  const float*     bn1b = (const float*)d_in[16];
  const float*     bn2g = (const float*)d_in[17];
  const float*     bn2b = (const float*)d_in[18];
  const float*     mW1 = (const float*)d_in[19];
  const float*     mb1 = (const float*)d_in[20];
  const float*     mW2 = (const float*)d_in[21];
  const float*     mb2 = (const float*)d_in[22];
  const float*     fcW = (const float*)d_in[23];
  const float*     fcb = (const float*)d_in[24];

  const int N = in_sizes[0] / 128;
  const int E = in_sizes[2] / 16;

  size_t off = 0;
  char* base = (char*)d_ws;
  auto alloc = [&](size_t bytes) -> char* {
    char* p = base + off;
    off += (bytes + 255) & ~(size_t)255;
    return p;
  };
  unsigned short* w1t  = (unsigned short*)alloc((size_t)256 * 128 * 2);
  unsigned short* w2t  = (unsigned short*)alloc((size_t)256 * 256 * 2);
  unsigned short* w3t  = (unsigned short*)alloc((size_t)64 * 256 * 2);
  unsigned short* mw1t = (unsigned short*)alloc((size_t)64 * 32 * 2);
  unsigned short* mw2t = (unsigned short*)alloc((size_t)64 * 64 * 2);
  unsigned short* abf  = (unsigned short*)alloc((size_t)N * 256 * 2);
  float* hfeat = (float*)alloc((size_t)N * 256 * 4);
  float* agg   = (float*)alloc((size_t)N * 256 * 4);
  float* lsb   = (float*)alloc((size_t)N * 4 * 4);
  float* ldb   = (float*)alloc((size_t)N * 4 * 4);
  float* mseg  = (float*)alloc((size_t)N * 4 * 4);
  float* sseg  = (float*)alloc((size_t)N * 4 * 4);
  float* h3    = (float*)alloc((size_t)N * 64 * 4);
  float* mu    = (float*)alloc(256 * 4);
  float* var   = (float*)alloc(256 * 4);

  auto cdiv = [](size_t a, size_t b) { return (unsigned int)((a + b - 1) / b); };

  // pack weights -> bf16 transposed
  pack_wt_k<<<cdiv(256 * 128, 256), 256, 0, stream>>>(W1, w1t, 128, 256, 128);
  pack_wt_k<<<cdiv(256 * 256, 256), 256, 0, stream>>>(W2, w2t, 256, 256, 256);
  pack_wt_k<<<cdiv(64 * 256, 256), 256, 0, stream>>>(W3, w3t, 256, 64, 256);
  pack_wt_k<<<cdiv(64 * 32, 256), 256, 0, stream>>>(mW1, mw1t, 16, 64, 32);
  pack_wt_k<<<cdiv(64 * 64, 256), 256, 0, stream>>>(mW2, mw2t, 64, 64, 64);
  f2bf_k<<<cdiv((size_t)N * 128, 256), 256, 0, stream>>>(x, abf, (size_t)N * 128);

  for (int L = 0; L < 3; ++L) {
    const int K  = (L == 0) ? 128 : 256;
    const int Nc = (L == 2) ? 64 : 256;
    const int H  = (L == 2) ? 1 : 4;
    const unsigned short* wt = (L == 0) ? w1t : (L == 1) ? w2t : w3t;
    const float* as_ = (L == 0) ? a1s : (L == 1) ? a2s : a3s;
    const float* ad_ = (L == 0) ? a1d : (L == 1) ? a2d : a3d;
    const float* bb  = (L == 0) ? b1 : (L == 1) ? b2 : b3;

    size_t tiles = (size_t)((N + 15) / 16) * (Nc / 16);
    wmma_gemm_k<<<cdiv(tiles, 8), 256, 0, stream>>>(abf, wt, hfeat, N, K, Nc);

    attn_k<<<cdiv((size_t)N * H, 8), 256, 0, stream>>>(hfeat, as_, ad_, lsb, ldb, N, H);

    fill_k<<<cdiv((size_t)N * H, 256), 256, 0, stream>>>(mseg, -__builtin_inff(), (size_t)N * H);
    fill_k<<<cdiv((size_t)N * H, 256), 256, 0, stream>>>(sseg, 0.0f, (size_t)N * H);
    fill_k<<<cdiv((size_t)N * Nc, 256), 256, 0, stream>>>(agg, 0.0f, (size_t)N * Nc);

    const int tot = E + N;
    edge_max_k<<<cdiv(tot, 256), 256, 0, stream>>>(ei, E, N, lsb, ldb, mseg, H);
    edge_expsum_k<<<cdiv(tot, 256), 256, 0, stream>>>(ei, E, N, lsb, ldb, mseg, sseg, H);
    size_t sthreads = (size_t)tot * H * 64;
    edge_scatter_k<<<cdiv(sthreads, 256), 256, 0, stream>>>(ei, E, N, lsb, ldb, mseg, sseg,
                                                            hfeat, agg, H);
    if (L < 2) {
      size_t total = (size_t)N * Nc;
      bias_act_k<<<cdiv(total, 256), 256, 0, stream>>>(agg, bb, agg, total, Nc, 1);
      bn_stats_k<<<Nc, 256, 0, stream>>>(agg, mu, var, N, Nc);
      const float* g_ = (L == 0) ? bn1g : bn2g;
      const float* bB = (L == 0) ? bn1b : bn2b;
      bn_apply_k<<<cdiv(total, 256), 256, 0, stream>>>(agg, mu, var, g_, bB, abf, total, Nc);
    } else {
      size_t total = (size_t)N * 64;
      bias_act_k<<<cdiv(total, 256), 256, 0, stream>>>(agg, bb, h3, total, 64, 0);
    }
  }

  final_edge_k<<<cdiv(E, 128), 256, 0, stream>>>(ei, E, ea, mw1t, mb1, mw2t, mb2,
                                                 h3, fcW, fcb, (float*)d_out);
  (void)n_in; (void)out_size; (void)ws_size;
}
